// ChunkRnn_71201967833917
// MI455X (gfx1250) — matrix operand order repaired and checked by reading, not compiled, not run on hardware
//
#include <hip/hip_runtime.h>
#include <hip/hip_bf16.h>
#include <stdint.h>

typedef unsigned short u16;
typedef __attribute__((ext_vector_type(16))) __bf16 v16bf;
typedef __attribute__((ext_vector_type(8)))  float  v8f;
typedef __attribute__((ext_vector_type(4)))  unsigned int v4u;
typedef __attribute__((ext_vector_type(4)))  int v4i;
typedef __attribute__((ext_vector_type(8)))  int v8i;

#define SEQ    4096
#define BATCH  2
#define MTOT   (BATCH * SEQ)   /* 8192 rows */

#if __has_builtin(__builtin_amdgcn_tensor_load_to_lds)
#define HAVE_TDM 1
#else
#define HAVE_TDM 0
#endif

// ---------------- helpers ----------------
__device__ __forceinline__ u16 f2bf(float f) {
  unsigned int u = __float_as_uint(f);
  u += 0x7FFFu + ((u >> 16) & 1u);        // round-to-nearest-even
  return (u16)(u >> 16);
}
__device__ __forceinline__ float bf2f(u16 h) {
  return __uint_as_float(((unsigned int)h) << 16);
}

struct BFrag { union { v16bf v; u16 u[16]; uint4 q[2]; }; };

__device__ __forceinline__ v8f vzero() {
  v8f z;
#pragma unroll
  for (int i = 0; i < 8; ++i) z[i] = 0.f;
  return z;
}

// A operand (16x32 bf16), row-major in LDS, stride ld (u16 units).
__device__ __forceinline__ void load_frag_a(const u16* base, int ld, int lane, BFrag& f) {
  const u16* row = base + (lane & 15) * ld + ((lane >> 4) << 3);
  f.q[0] = *(const uint4*)(row);
  f.q[1] = *(const uint4*)(row + 16);
}
// B operand (32x16 bf16) stored TRANSPOSED in LDS as [n][k], stride ld.
__device__ __forceinline__ void load_frag_b(const u16* baseT, int ld, int lane, BFrag& f) {
  const u16* row = baseT + (lane & 15) * ld + ((lane >> 4) << 3);
  f.q[0] = *(const uint4*)(row);
  f.q[1] = *(const uint4*)(row + 16);
}
__device__ __forceinline__ v8f wmma_bf16(const BFrag& a, const BFrag& b, v8f c) {
  return __builtin_amdgcn_wmma_f32_16x16x32_bf16(false, a.v, false, b.v, (short)0, c,
                                                 false, false);
}

#if HAVE_TDM
// Issue one 2D TDM tile load: tile_d1 rows of tile_d0 elements, global row
// stride stride_d0 elements, into LDS at lds_addr with optional row padding.
// D# layout per CDNA5 ISA (group0 128b, group1 256b).
__device__ __forceinline__ void tdm_load_tile(
    unsigned lds_addr, const void* gptr,
    unsigned tensor_d0, unsigned tile_d0, unsigned tile_d1,
    unsigned stride_d0, unsigned ds_code /*1=2B,2=4B*/,
    unsigned pad_int, unsigned pad_amt)
{
  unsigned long long ga = (unsigned long long)(uintptr_t)gptr;
  v4u g0;
  g0[0] = 1u;                                                  // count=1, user D#
  g0[1] = lds_addr;                                            // LDS byte address
  g0[2] = (unsigned)ga;                                        // global_addr[31:0]
  g0[3] = (unsigned)((ga >> 32) & 0x01FFFFFFu) | (2u << 30);   // [56:32] | type=2
  v8i g1;
  g1[0] = (int)((ds_code << 16) | (1u << 20) |                 // data_size | pad_enable
                (pad_int << 22) | (pad_amt << 25));            // pad codes
  g1[1] = (int)((tensor_d0 & 0xFFFFu) << 16);                  // tensor_dim0 lo16
  g1[2] = (int)((tensor_d0 >> 16) & 0xFFFFu);                  // tensor_dim0 hi16 | dim1 lo16=0
  g1[3] = (int)(0x10u | (tile_d0 << 16));                      // dim1 hi16 (dim1=1M) | tile_dim0
  g1[4] = (int)(tile_d1 & 0xFFFFu);                            // tile_dim1 | tile_dim2=0
  g1[5] = (int)stride_d0;                                      // tensor_dim0_stride lo32
  g1[6] = 0;                                                   // stride hi | dim1_stride lo
  g1[7] = 0;
  v4i gz; gz[0] = gz[1] = gz[2] = gz[3] = 0;
#if defined(__clang_major__) && (__clang_major__ >= 23)
  v8i gz8;
#pragma unroll
  for (int i = 0; i < 8; ++i) gz8[i] = 0;
  __builtin_amdgcn_tensor_load_to_lds(g0, g1, gz, gz, gz8, 0);
#else
  __builtin_amdgcn_tensor_load_to_lds(g0, g1, gz, gz, 0);
#endif
}
#endif

// ================= kernel 1: proj GEMM + activations =================
__global__ __launch_bounds__(256) void proj_act_kernel(
    const float* __restrict__ X, const float* __restrict__ W,
    u16* __restrict__ qh, u16* __restrict__ kh, u16* __restrict__ vh,
    float* __restrict__ lfb)
{
  __shared__ __align__(16) u16 As[2][128][40];   // [m][k] bf16, double buffered
  __shared__ __align__(16) u16 Bs[2][128][40];   // [n][k] bf16 (transposed)
  const int tid  = threadIdx.x;
  const int lane = tid & 31;
  const int wave = tid >> 5;
  const int m0 = blockIdx.y * 128;
  const int n0 = blockIdx.x * 128;
  const int wm = wave & 3;
  const int wn = wave >> 2;

  v8f acc[2][4];
#pragma unroll
  for (int i = 0; i < 2; ++i)
#pragma unroll
    for (int j = 0; j < 4; ++j) acc[i][j] = vzero();

  auto stage = [&](int k0, int buf) {
    float4 xa[4], wb[4];
#pragma unroll
    for (int i = 0; i < 4; ++i) {                // batch all global loads first
      int l = tid + i * 256;
      xa[i] = *(const float4*)(X + (size_t)(m0 + (l >> 3)) * 1024 + k0 + ((l & 7) << 2));
      wb[i] = *(const float4*)(W + (size_t)(k0 + (l >> 5)) * 3072 + n0 + ((l & 31) << 2));
    }
#pragma unroll
    for (int i = 0; i < 4; ++i) {                // then convert + LDS store
      int l = tid + i * 256;
      int row = l >> 3, c4 = (l & 7) << 2;
      u16* d = &As[buf][row][c4];
      d[0] = f2bf(xa[i].x); d[1] = f2bf(xa[i].y); d[2] = f2bf(xa[i].z); d[3] = f2bf(xa[i].w);
      int k = l >> 5, n4 = (l & 31) << 2;
      Bs[buf][n4 + 0][k] = f2bf(wb[i].x); Bs[buf][n4 + 1][k] = f2bf(wb[i].y);
      Bs[buf][n4 + 2][k] = f2bf(wb[i].z); Bs[buf][n4 + 3][k] = f2bf(wb[i].w);
    }
  };

  stage(0, 0);
  __syncthreads();
  for (int kb = 0; kb < 32; ++kb) {
    int cur = kb & 1;
    if (kb + 1 < 32) stage((kb + 1) * 32, cur ^ 1);   // overlap with WMMA below
    if (kb + 2 < 32) {                                // gfx1250 global_prefetch_b8
      __builtin_prefetch((const void*)(X + (size_t)(m0 + (tid >> 1)) * 1024 + (kb + 2) * 32), 0, 1);
      __builtin_prefetch((const void*)(W + (size_t)((kb + 2) * 32 + (tid >> 3)) * 3072 + n0), 0, 1);
    }
    BFrag a[2], b[4];
#pragma unroll
    for (int i = 0; i < 2; ++i) load_frag_a(&As[cur][wm * 32 + i * 16][0], 40, lane, a[i]);
#pragma unroll
    for (int j = 0; j < 4; ++j) load_frag_b(&Bs[cur][wn * 64 + j * 16][0], 40, lane, b[j]);
#pragma unroll
    for (int i = 0; i < 2; ++i)
#pragma unroll
      for (int j = 0; j < 4; ++j) acc[i][j] = wmma_bf16(a[i], b[j], acc[i][j]);
    __syncthreads();   // double buffer: one barrier per iteration
  }

  const int seg = n0 >> 10;                     // 0:q  1:gate  2:v
  const int cn = lane & 15;
  const int rb = (lane >> 4) << 3;              // C layout: M = r + 8*(lane>=16)
#pragma unroll
  for (int i = 0; i < 2; ++i)
#pragma unroll
    for (int j = 0; j < 4; ++j)
#pragma unroll
      for (int r = 0; r < 8; ++r) {
        int gm = m0 + wm * 32 + i * 16 + rb + r;
        int lc = (n0 + wn * 64 + j * 16 + cn) - seg * 1024;
        float val = acc[i][j][r];
        size_t idx = (size_t)gm * 1024 + lc;
        if (seg == 0) {                          // q = silu(q)
          float s = 1.f / (1.f + __expf(-val));
          qh[idx] = f2bf(val * s);
        } else if (seg == 1) {                   // f=sigmoid; log_f=log(sigmoid(f)); k=1-f
          float f = 1.f / (1.f + __expf(-val));
          lfb[idx] = -__logf(1.f + __expf(-f));
          kh[idx] = f2bf(1.f - f);
        } else {
          vh[idx] = f2bf(val);
        }
      }
}

// ================= kernel 2: chunked GLA scan =================
__global__ __launch_bounds__(256) void chunk_scan_kernel(
    const u16* __restrict__ qh, const u16* __restrict__ kh,
    const u16* __restrict__ vh, const float* __restrict__ lfb,
    float* __restrict__ oat)
{
  __shared__ __align__(16) float bcs[64][65];   // log_f -> cumsum
  __shared__ __align__(16) u16 qe[64][72];      // q*exp(bc)      [t][e]
  __shared__ __align__(16) u16 kd[64][72];      // k*exp(-bc)     [t][e]
  __shared__ __align__(16) u16 kdT[64][72];     //                [e][t]
  __shared__ __align__(16) u16 vsT[64][72];     // v transposed   [e2][t]
  __shared__ __align__(16) u16 Am[64][72];      // tril attention [i][j]
  __shared__ __align__(16) float Sf[64][65];    // state f32      [e][e2]
  __shared__ __align__(16) u16 SbT[64][72];     // state bf16     [e2][e]
  __shared__ float eb[64];                      // exp(bc_last)

  const int tid = threadIdx.x;
  const int lane = tid & 31;
  const int wave = tid >> 5;
  const int b = blockIdx.x >> 4;
  const int h = blockIdx.x & 15;
  const int r  = tid >> 2;             // token row 0..63
  const int e0 = (tid & 3) << 4;       // channel group
  const int cn = lane & 15;
  const int rb = (lane >> 4) << 3;

  for (int idx = tid; idx < 64 * 65; idx += 256) ((float*)Sf)[idx] = 0.f;
  for (int idx = tid; idx < 64 * 72; idx += 256) { ((u16*)SbT)[idx] = 0; ((u16*)Am)[idx] = 0; }
  __syncthreads();

  static const signed char TI[10] = {0,1,1,2,2,2,3,3,3,3};
  static const signed char TJ[10] = {0,0,1,0,1,2,0,1,2,3};

  for (int c = 0; c < 64; ++c) {
    // ---- stage q,k,lf via Tensor Data Mover; v manually (needs transpose) ----
#if HAVE_TDM
    if (wave == 0) {
      size_t goff = ((size_t)(b * SEQ + c * 64)) * 1024 + h * 64;
      // bf16 tiles: 64x64, row 128B, pad 16B -> 72-u16 LDS stride (codes: 32dw interval, 4dw pad)
      tdm_load_tile((unsigned)(uintptr_t)(void*)&qe[0][0],  qh  + goff, 1024, 64, 64, 1024, 1, 4, 3);
      tdm_load_tile((unsigned)(uintptr_t)(void*)&kd[0][0],  kh  + goff, 1024, 64, 64, 1024, 1, 4, 3);
      // f32 tile: row 256B, pad 4B -> 65-f32 LDS stride (codes: 64dw interval, 1dw pad)
      tdm_load_tile((unsigned)(uintptr_t)(void*)&bcs[0][0], lfb + goff, 1024, 64, 64, 1024, 2, 5, 0);
    }
#endif
    {
      size_t g = ((size_t)(b * SEQ + c * 64 + r)) * 1024 + h * 64 + e0;
#if !HAVE_TDM
      *(uint4*)&qe[r][e0]     = *(const uint4*)(qh + g);
      *(uint4*)&qe[r][e0 + 8] = *(const uint4*)(qh + g + 8);
      *(uint4*)&kd[r][e0]     = *(const uint4*)(kh + g);
      *(uint4*)&kd[r][e0 + 8] = *(const uint4*)(kh + g + 8);
      const float4* lp = (const float4*)(lfb + g);
#pragma unroll
      for (int j4 = 0; j4 < 4; ++j4) *(float4*)&bcs[r][e0 + 4 * j4] = lp[j4];
#endif
      uint4 v0 = *(const uint4*)(vh + g), v1 = *(const uint4*)(vh + g + 8);
      u16 tmp[16];
      *(uint4*)&tmp[0] = v0; *(uint4*)&tmp[8] = v1;
#pragma unroll
      for (int j = 0; j < 16; ++j) vsT[e0 + j][r] = tmp[j];   // transpose into LDS
    }
#if HAVE_TDM
    if (wave == 0) __builtin_amdgcn_s_wait_tensorcnt(0);
#endif
    __syncthreads();
    if (tid < 64) {                    // ---- cumsum over t per channel ----
      float run = 0.f;
      for (int t = 0; t < 64; ++t) { run += bcs[t][tid]; bcs[t][tid] = run; }
      eb[tid] = __expf(run);
    }
    __syncthreads();
#pragma unroll
    for (int j = 0; j < 16; ++j) {     // ---- decay-scaled operands ----
      int e = e0 + j;
      float bcv = bcs[r][e];
      qe[r][e] = f2bf(bf2f(qe[r][e]) * __expf(bcv));
      u16 kv = f2bf(bf2f(kd[r][e]) * __expf(-bcv));
      kd[r][e] = kv;
      kdT[e][r] = kv;
    }
    __syncthreads();
    // ---- A = tril(qe @ kd^T): 10 lower-triangular 16x16 tiles ----
    for (int p = wave; p < 10; p += 8) {
      int I = TI[p], J = TJ[p];
      v8f acc = vzero();
#pragma unroll
      for (int ks = 0; ks < 64; ks += 32) {
        BFrag a, bb;
        load_frag_a(&qe[I * 16][ks], 72, lane, a);
        load_frag_b(&kd[J * 16][ks], 72, lane, bb);
        acc = wmma_bf16(a, bb, acc);
      }
#pragma unroll
      for (int rr = 0; rr < 8; ++rr) {
        int ri = rb + rr;
        float v = acc[rr];
        if (I == J && cn > ri) v = 0.f;               // causal mask
        Am[I * 16 + ri][J * 16 + cn] = f2bf(v);
      }
    }
    __syncthreads();
    // ---- o = Am @ v + qe @ S ----
#pragma unroll
    for (int s2 = 0; s2 < 2; ++s2) {
      int t2 = wave * 2 + s2;
      int I = t2 >> 2, J = t2 & 3;
      v8f acc = vzero();
#pragma unroll
      for (int ks = 0; ks < 64; ks += 32) {
        BFrag a, bb;
        load_frag_a(&Am[I * 16][ks], 72, lane, a);
        load_frag_b(&vsT[J * 16][ks], 72, lane, bb);
        acc = wmma_bf16(a, bb, acc);
      }
#pragma unroll
      for (int ks = 0; ks < 64; ks += 32) {
        BFrag a, bb;
        load_frag_a(&qe[I * 16][ks], 72, lane, a);
        load_frag_b(&SbT[J * 16][ks], 72, lane, bb);
        acc = wmma_bf16(a, bb, acc);
      }
#pragma unroll
      for (int rr = 0; rr < 8; ++rr) {
        int t = c * 64 + I * 16 + rb + rr;
        oat[((size_t)(b * SEQ + t)) * 1024 + h * 64 + J * 16 + cn] = acc[rr];
      }
    }
    __syncthreads();
    // ---- S = exp(bc_last) * (S + kd^T @ v) ----
#pragma unroll
    for (int s2 = 0; s2 < 2; ++s2) {
      int t2 = wave * 2 + s2;
      int I = t2 >> 2, J = t2 & 3;
      v8f acc = vzero();
#pragma unroll
      for (int ks = 0; ks < 64; ks += 32) {
        BFrag a, bb;
        load_frag_a(&kdT[I * 16][ks], 72, lane, a);
        load_frag_b(&vsT[J * 16][ks], 72, lane, bb);
        acc = wmma_bf16(a, bb, acc);
      }
#pragma unroll
      for (int rr = 0; rr < 8; ++rr) {
        int e  = I * 16 + rb + rr;
        int e2 = J * 16 + cn;
        float sv = eb[e] * (Sf[e][e2] + acc[rr]);
        Sf[e][e2] = sv;
        SbT[e2][e] = f2bf(sv);
      }
    }
    __syncthreads();
  }
}

// ================= kernel 3: LayerNorm -> bf16 =================
__global__ __launch_bounds__(256) void ln_kernel(
    const float* __restrict__ oat, const float* __restrict__ nw,
    u16* __restrict__ onorm)
{
  __shared__ float rs[256], rq[256];
  __shared__ float smu, srsg;
  const int row = blockIdx.x, tid = threadIdx.x;
  const float* o = oat + (size_t)row * 1024;
  float s = 0.f, q = 0.f;
  for (int j = tid; j < 1024; j += 256) { float v = o[j]; s += v; q += v * v; }
  rs[tid] = s; rq[tid] = q;
  __syncthreads();
  for (int off = 128; off > 0; off >>= 1) {
    if (tid < off) { rs[tid] += rs[tid + off]; rq[tid] += rq[tid + off]; }
    __syncthreads();
  }
  if (tid == 0) {
    float mu = rs[0] * (1.f / 1024.f);
    float var = rq[0] * (1.f / 1024.f) - mu * mu;
    smu = mu; srsg = rsqrtf(var + 1e-5f);
  }
  __syncthreads();
  float mu = smu, rsg = srsg;
  for (int j = tid; j < 1024; j += 256)
    onorm[(size_t)row * 1024 + j] = f2bf((o[j] - mu) * rsg * nw[j]);
}

// ================= kernel 4: output GEMM =================
__global__ __launch_bounds__(256) void out_gemm_kernel(
    const u16* __restrict__ Ag, const float* __restrict__ W,
    float* __restrict__ out)
{
  __shared__ __align__(16) u16 As[2][128][40];
  __shared__ __align__(16) u16 Bs[2][128][40];
  const int tid = threadIdx.x, lane = tid & 31, wave = tid >> 5;
  const int m0 = blockIdx.y * 128, n0 = blockIdx.x * 128;
  const int wm = wave & 3, wn = wave >> 2;

  v8f acc[2][4];
#pragma unroll
  for (int i = 0; i < 2; ++i)
#pragma unroll
    for (int j = 0; j < 4; ++j) acc[i][j] = vzero();

  auto stage = [&](int k0, int buf) {
    uint4 av[2]; float4 wb[4];
#pragma unroll
    for (int i = 0; i < 2; ++i) {
      int l = tid + i * 256;
      av[i] = *(const uint4*)(Ag + (size_t)(m0 + (l >> 2)) * 1024 + k0 + ((l & 3) << 3));
    }
#pragma unroll
    for (int i = 0; i < 4; ++i) {
      int l = tid + i * 256;
      wb[i] = *(const float4*)(W + (size_t)(k0 + (l >> 5)) * 1024 + n0 + ((l & 31) << 2));
    }
#pragma unroll
    for (int i = 0; i < 2; ++i) {
      int l = tid + i * 256;
      *(uint4*)&As[buf][l >> 2][(l & 3) << 3] = av[i];
    }
#pragma unroll
    for (int i = 0; i < 4; ++i) {
      int l = tid + i * 256;
      int k = l >> 5, n4 = (l & 31) << 2;
      Bs[buf][n4 + 0][k] = f2bf(wb[i].x); Bs[buf][n4 + 1][k] = f2bf(wb[i].y);
      Bs[buf][n4 + 2][k] = f2bf(wb[i].z); Bs[buf][n4 + 3][k] = f2bf(wb[i].w);
    }
  };

  stage(0, 0);
  __syncthreads();
  for (int kb = 0; kb < 32; ++kb) {
    int cur = kb & 1;
    if (kb + 1 < 32) stage((kb + 1) * 32, cur ^ 1);
    if (kb + 2 < 32)
      __builtin_prefetch((const void*)(W + (size_t)((kb + 2) * 32 + (tid >> 3)) * 1024 + n0), 0, 1);
    BFrag a[2], b[4];
#pragma unroll
    for (int i = 0; i < 2; ++i) load_frag_a(&As[cur][wm * 32 + i * 16][0], 40, lane, a[i]);
#pragma unroll
    for (int j = 0; j < 4; ++j) load_frag_b(&Bs[cur][wn * 64 + j * 16][0], 40, lane, b[j]);
#pragma unroll
    for (int i = 0; i < 2; ++i)
#pragma unroll
      for (int j = 0; j < 4; ++j) acc[i][j] = wmma_bf16(a[i], b[j], acc[i][j]);
    __syncthreads();
  }
  const int cn = lane & 15, rb = (lane >> 4) << 3;
#pragma unroll
  for (int i = 0; i < 2; ++i)
#pragma unroll
    for (int j = 0; j < 4; ++j)
#pragma unroll
      for (int r = 0; r < 8; ++r) {
        int gm = m0 + wm * 32 + i * 16 + rb + r;
        int gn = n0 + wn * 64 + j * 16 + cn;
        out[(size_t)gm * 1024 + gn] = acc[i][j][r];
      }
}

// ================= host launcher =================
extern "C" void kernel_launch(void* const* d_in, const int* in_sizes, int n_in,
                              void* d_out, int out_size, void* d_ws, size_t ws_size,
                              hipStream_t stream) {
  const float* x     = (const float*)d_in[0];
  const float* W_in  = (const float*)d_in[1];
  const float* W_out = (const float*)d_in[2];
  const float* nw    = (const float*)d_in[3];
  float* out = (float*)d_out;

  const size_t MN = (size_t)MTOT * 1024;
  char* w = (char*)d_ws;
  u16*   qh    = (u16*)w;   w += MN * 2;
  u16*   kh    = (u16*)w;   w += MN * 2;
  u16*   vh    = (u16*)w;   w += MN * 2;
  float* lfb   = (float*)w; w += MN * 4;
  float* oat   = (float*)w; w += MN * 4;
  u16*   onorm = (u16*)w;   w += MN * 2;        // 128 MB total

  proj_act_kernel<<<dim3(3072 / 128, MTOT / 128), 256, 0, stream>>>(x, W_in, qh, kh, vh, lfb);
  chunk_scan_kernel<<<dim3(BATCH * 16), 256, 0, stream>>>(qh, kh, vh, lfb, oat);
  ln_kernel<<<dim3(MTOT), 256, 0, stream>>>(oat, nw, onorm);
  out_gemm_kernel<<<dim3(1024 / 128, MTOT / 128), 256, 0, stream>>>(onorm, W_out, out);
}
